// GNN_88648124990605
// MI455X (gfx1250) — compile-verified, matrix operand order checked by reading, not run on hardware
//
#include <hip/hip_runtime.h>
#include <hip/hip_bf16.h>

typedef __attribute__((ext_vector_type(16))) __bf16 v16bf;
typedef __attribute__((ext_vector_type(8)))  __bf16 v8bf;
typedef __attribute__((ext_vector_type(8)))  float  v8f;

#define N_NODES  50000
#define N_EDGES  800000
#define N_GRAPHS 512
#define EMB      64
#define HID      128

static __device__ __forceinline__ void atomAddF(float* p, float v) {
    __hip_atomic_fetch_add(p, v, __ATOMIC_RELAXED, __HIP_MEMORY_SCOPE_AGENT);
}

// ---------------- embedding gather: h0[n, 0:64] = emb_table[x[n]] ----------------
__global__ void k_embed(const int* __restrict__ x, const float* __restrict__ tab,
                        float* __restrict__ h0) {
    int tid = blockIdx.x * blockDim.x + threadIdx.x;        // N*16 float4 slots
    if (tid >= N_NODES * 16) return;
    int n = tid >> 4, q = tid & 15;
    ((float4*)h0)[tid] = ((const float4*)tab)[(long)x[n] * 16 + q];
}

// ---------------- degree: deg[dst] += 1 ----------------
__global__ void k_deg(const int* __restrict__ dstIdx, float* __restrict__ deg) {
    int e = blockIdx.x * blockDim.x + threadIdx.x;
    if (e < N_EDGES) atomAddF(&deg[dstIdx[e]], 1.0f);
}

// ---------------- edge scatter-add: agg[dst, :] += h[src, :] (F = 4*Fq) ----------------
__global__ void k_scatter(const int* __restrict__ srcIdx, const int* __restrict__ dstIdx,
                          const float* __restrict__ h, float* __restrict__ agg, int lq) {
    int tid = blockIdx.x * blockDim.x + threadIdx.x;        // E * Fq threads
    int fq  = 1 << lq;
    if (tid >= N_EDGES * fq) return;
    int e = tid >> lq, q = tid & (fq - 1);
    int s = srcIdx[e], d = dstIdx[e];
    float4 v = ((const float4*)h)[(long)s * fq + q];
    float* ap = agg + ((long)d * fq + q) * 4;
    atomAddF(ap + 0, v.x); atomAddF(ap + 1, v.y);
    atomAddF(ap + 2, v.z); atomAddF(ap + 3, v.w);
}

// ---------------- prep: meanbf = bf16(agg/deg), hbf = bf16(h) ----------------
__global__ void k_prep(const float* __restrict__ agg, const float* __restrict__ h,
                       const float* __restrict__ deg,
                       __bf16* __restrict__ meanbf, __bf16* __restrict__ hbf, int lf) {
    int tid = blockIdx.x * blockDim.x + threadIdx.x;        // N * F
    if (tid >= (N_NODES << lf)) return;
    int n = tid >> lf;
    float r = 1.0f / fmaxf(deg[n], 1.0f);
    meanbf[tid] = (__bf16)(agg[tid] * r);
    hbf[tid]    = (__bf16)h[tid];
}

// ---------------- pack W[K,128] fp32 -> bf16 WMMA B-fragment order ----------------
// frag layout: [n_tile][k_step][lane][16 elems] ; lane = 16*half + n%16
// elem i<8 : k = 32*step + 8*half + i ; elem i>=8 : k = 32*step + 16 + 8*half + (i-8)
__global__ void k_pack(const float* __restrict__ W, __bf16* __restrict__ Wp, int nsteps) {
    int tid = blockIdx.x * blockDim.x + threadIdx.x;
    int total = 8 * nsteps * 512;
    if (tid >= total) return;
    int i      = tid & 15;
    int lane   = (tid >> 4) & 31;
    int step   = (tid >> 9) % nsteps;
    int n_tile = tid / (512 * nsteps);
    int half   = lane >> 4;
    int n      = n_tile * 16 + (lane & 15);
    int klocal = (i < 8) ? (8 * half + i) : (16 + 8 * half + (i - 8));
    int k      = step * 32 + klocal;
    Wp[tid] = (__bf16)W[k * HID + n];
}

// ---------------- fused SAGE GEMM: out = relu(mean@Wl + h@Wr + bias), [N,K]x[K,128] ----------------
__global__ __launch_bounds__(256)
void k_gemm(const __bf16* __restrict__ Amean, const __bf16* __restrict__ Aself,
            const __bf16* __restrict__ Wl, const __bf16* __restrict__ Wr,
            const float* __restrict__ bias, float* __restrict__ out, int K) {
    int lane   = threadIdx.x & 31;
    int n_tile = threadIdx.x >> 5;                 // 8 waves -> 8 column tiles
    int m_tile = blockIdx.x;                       // 3125 row tiles
    int nsteps = K >> 5;
    int half   = lane >> 4;
    int m      = m_tile * 16 + (lane & 15);
    int n      = n_tile * 16 + (lane & 15);

    v8f acc = {0.f, 0.f, 0.f, 0.f, 0.f, 0.f, 0.f, 0.f};

    for (int pass = 0; pass < 2; ++pass) {
        const __bf16* A  = pass ? Aself : Amean;
        const __bf16* Wp = pass ? Wr : Wl;
        const __bf16* arow  = A + (long)m * K + half * 8;
        const __bf16* bbase = Wp + ((long)n_tile * nsteps * 32 + lane) * 16;
        for (int s = 0; s < nsteps; ++s) {
            v8bf alo = *(const v8bf*)(arow + s * 32);        // k = 32s + 8*half + [0..7]
            v8bf ahi = *(const v8bf*)(arow + s * 32 + 16);   // k = 32s + 16 + 8*half + [0..7]
            v16bf a;
#pragma unroll
            for (int i = 0; i < 8; ++i) { a[i] = alo[i]; a[i + 8] = ahi[i]; }
            v16bf b = *(const v16bf*)(bbase + (long)s * 512);
            acc = __builtin_amdgcn_wmma_f32_16x16x32_bf16(
                      false, a, false, b, (short)0, acc, false, false);
        }
    }

    float bv    = bias[n];
    int   mbase = m_tile * 16 + half * 8;
#pragma unroll
    for (int r = 0; r < 8; ++r) {
        float v = acc[r] + bv;
        out[(long)(mbase + r) * HID + n] = v > 0.f ? v : 0.f;
    }
}

// ---------------- pooling ----------------
__global__ void k_pool(const float* __restrict__ h2, const int* __restrict__ batch,
                       float* __restrict__ pooled) {
    int tid = blockIdx.x * blockDim.x + threadIdx.x;        // N * 32 float4 slots
    if (tid >= N_NODES * 32) return;
    int n = tid >> 5, q = tid & 31;
    int g = batch[n];
    float4 v = ((const float4*)h2)[(long)n * 32 + q];
    float* p = pooled + ((long)g * 32 + q) * 4;
    atomAddF(p + 0, v.x); atomAddF(p + 1, v.y);
    atomAddF(p + 2, v.z); atomAddF(p + 3, v.w);
}

__global__ void k_cnt(const int* __restrict__ batch, float* __restrict__ cnt) {
    int n = blockIdx.x * blockDim.x + threadIdx.x;
    if (n < N_NODES) atomAddF(&cnt[batch[n]], 1.0f);
}

__global__ void k_head(const float* __restrict__ pooled, const float* __restrict__ cnt,
                       const float* __restrict__ wo, const float* __restrict__ bo,
                       float* __restrict__ out) {
    int tid = blockIdx.x * blockDim.x + threadIdx.x;        // G*2
    if (tid >= N_GRAPHS * 2) return;
    int g = tid >> 1, c = tid & 1;
    float inv = 1.0f / fmaxf(cnt[g], 1.0f);
    float s = 0.f;
    for (int t = 0; t < HID; ++t) s += pooled[g * HID + t] * wo[t * 2 + c];
    out[tid] = s * inv + bo[c];
}

extern "C" void kernel_launch(void* const* d_in, const int* in_sizes, int n_in,
                              void* d_out, int out_size, void* d_ws, size_t ws_size,
                              hipStream_t stream) {
    const int*   x     = (const int*)d_in[0];
    const int*   srcI  = (const int*)d_in[1];
    const int*   dstI  = srcI + N_EDGES;
    const int*   batch = (const int*)d_in[2];
    const float* tab   = (const float*)d_in[3];
    const float* w1l   = (const float*)d_in[4];
    const float* b1    = (const float*)d_in[5];
    const float* w1r   = (const float*)d_in[6];
    const float* w2l   = (const float*)d_in[7];
    const float* b2    = (const float*)d_in[8];
    const float* w2r   = (const float*)d_in[9];
    const float* wo    = (const float*)d_in[10];
    const float* bo    = (const float*)d_in[11];
    float* out = (float*)d_out;

    char* ws = (char*)d_ws;
    size_t off = 0;
    auto alloc = [&](size_t bytes) -> char* {
        char* p = ws + off;
        off += (bytes + 255) & ~(size_t)255;
        return p;
    };
    float*  h0     = (float*)alloc((size_t)N_NODES * EMB * 4);
    float*  agg1   = (float*)alloc((size_t)N_NODES * EMB * 4);
    float*  deg    = (float*)alloc((size_t)N_NODES * 4);
    __bf16* h0bf   = (__bf16*)alloc((size_t)N_NODES * EMB * 2);
    __bf16* mean1  = (__bf16*)alloc((size_t)N_NODES * EMB * 2);
    float*  h1     = (float*)alloc((size_t)N_NODES * HID * 4);
    float*  agg2   = (float*)alloc((size_t)N_NODES * HID * 4);
    __bf16* h1bf   = (__bf16*)alloc((size_t)N_NODES * HID * 2);
    __bf16* mean2  = (__bf16*)alloc((size_t)N_NODES * HID * 2);
    float*  h2     = (float*)alloc((size_t)N_NODES * HID * 4);
    float*  pooled = (float*)alloc((size_t)N_GRAPHS * HID * 4);
    float*  cnt    = (float*)alloc((size_t)N_GRAPHS * 4);
    __bf16* w1lp   = (__bf16*)alloc((size_t)EMB * HID * 2);
    __bf16* w1rp   = (__bf16*)alloc((size_t)EMB * HID * 2);
    __bf16* w2lp   = (__bf16*)alloc((size_t)HID * HID * 2);
    __bf16* w2rp   = (__bf16*)alloc((size_t)HID * HID * 2);

    // zero all accumulation buffers (fresh each call — deterministic)
    hipMemsetAsync(agg1, 0, (size_t)N_NODES * EMB * 4, stream);
    hipMemsetAsync(deg,  0, (size_t)N_NODES * 4, stream);
    hipMemsetAsync(agg2, 0, (size_t)N_NODES * HID * 4, stream);
    hipMemsetAsync(pooled, 0, (size_t)N_GRAPHS * HID * 4, stream);
    hipMemsetAsync(cnt,  0, (size_t)N_GRAPHS * 4, stream);

    const int B = 256;
    // embedding gather
    k_embed<<<(N_NODES * 16 + B - 1) / B, B, 0, stream>>>(x, tab, h0);
    // weight packing (fragment order, bf16)
    k_pack<<<(8 * 2 * 512 + B - 1) / B, B, 0, stream>>>(w1l, w1lp, 2);
    k_pack<<<(8 * 2 * 512 + B - 1) / B, B, 0, stream>>>(w1r, w1rp, 2);
    k_pack<<<(8 * 4 * 512 + B - 1) / B, B, 0, stream>>>(w2l, w2lp, 4);
    k_pack<<<(8 * 4 * 512 + B - 1) / B, B, 0, stream>>>(w2r, w2rp, 4);
    // degrees (shared by both layers)
    k_deg<<<(N_EDGES + B - 1) / B, B, 0, stream>>>(dstI, deg);

    // ---- layer 1 ----
    k_scatter<<<(N_EDGES * 16 + B - 1) / B, B, 0, stream>>>(srcI, dstI, h0, agg1, 4);
    k_prep<<<((N_NODES * EMB) + B - 1) / B, B, 0, stream>>>(agg1, h0, deg, mean1, h0bf, 6);
    k_gemm<<<N_NODES / 16, B, 0, stream>>>(mean1, h0bf, w1lp, w1rp, b1, h1, EMB);

    // ---- layer 2 ----
    k_scatter<<<(N_EDGES * 32 + B - 1) / B, B, 0, stream>>>(srcI, dstI, h1, agg2, 5);
    k_prep<<<((N_NODES * HID) + B - 1) / B, B, 0, stream>>>(agg2, h1, deg, mean2, h1bf, 7);
    k_gemm<<<N_NODES / 16, B, 0, stream>>>(mean2, h1bf, w2lp, w2rp, b2, h2, HID);

    // ---- pool + head ----
    k_pool<<<(N_NODES * 32 + B - 1) / B, B, 0, stream>>>(h2, batch, pooled);
    k_cnt<<<(N_NODES + B - 1) / B, B, 0, stream>>>(batch, cnt);
    k_head<<<(N_GRAPHS * 2 + B - 1) / B, B, 0, stream>>>(pooled, cnt, wo, bo, out);
}
